// GraphNetworkBlock_20246475833428
// MI455X (gfx1250) — compile-verified, matrix operand order checked by reading, not run on hardware
//
#include <hip/hip_runtime.h>
#include <hip/hip_bf16.h>

// GraphNetworkBlock for MI455X (gfx1250, wave32, WMMA).
// Edge MLP + scatter-add + node MLP via v_wmma_f32_16x16x32_f16.
// Register blocking: each wave owns 2 M-tiles (32 rows) so every LDS B-fragment
// feeds 2 WMMAs -> halves DS traffic; B fragments batched 4-at-a-time so the
// ds_load clauses and s_wait_dscnt amortize over 8 back-to-back WMMAs.

#define LATENT   64
#define HIDDEN   128
#define N_NODES  100000
#define N_EDGES  1600000

typedef __attribute__((ext_vector_type(16))) _Float16 v16h;
typedef __attribute__((ext_vector_type(8)))  _Float16 v8h;
typedef __attribute__((ext_vector_type(8)))  float    v8f;
typedef __attribute__((ext_vector_type(4)))  float    f32x4;

// ---- WMMA fragment builders -------------------------------------------------
// A-matrix 16x32 f16 layout (ISA 7.12.2): lane l: m = l&15, h = l>>4.
// h=0 lanes hold K = {0..7, 16..23}; h=1 lanes hold K = {8..15, 24..31}.
// B-matrix 32x16 symmetric with n = l&15 (weights stored transposed W[n][k] in
// LDS so each half-fragment is one 16B ds_load_b128).

__device__ __forceinline__ v16h frag_from_global_f32(const float* __restrict__ rowp,
                                                     int koff, int h) {
    const float* q = rowp + koff + h * 8;
    f32x4 f0 = *(const f32x4*)(q);
    f32x4 f1 = *(const f32x4*)(q + 4);
    f32x4 f2 = *(const f32x4*)(q + 16);
    f32x4 f3 = *(const f32x4*)(q + 20);
    v16h a;
#pragma unroll
    for (int j = 0; j < 4; ++j) {
        a[j]      = (_Float16)f0[j];
        a[4 + j]  = (_Float16)f1[j];
        a[8 + j]  = (_Float16)f2[j];
        a[12 + j] = (_Float16)f3[j];
    }
    return a;
}

__device__ __forceinline__ v16h frag_from_lds(const _Float16* p, int stride,
                                              int rown, int koff, int h) {
    const _Float16* q = p + rown * stride + koff + h * 8;
    v8h lo = *(const v8h*)(q);
    v8h hi = *(const v8h*)(q + 16);
    v16h f;
#pragma unroll
    for (int j = 0; j < 8; ++j) { f[j] = lo[j]; f[8 + j] = hi[j]; }
    return f;
}

#define WMMA_F16(A, B, C) \
    __builtin_amdgcn_wmma_f32_16x16x32_f16(false, (A), false, (B), (short)0, (C), false, false)

// ---- workspace zeroing ------------------------------------------------------
__global__ void zero_kernel(float* __restrict__ p, int n) {
    int i = blockIdx.x * blockDim.x + threadIdx.x;
    if (i < n) p[i] = 0.0f;
}

// ---- edge kernel ------------------------------------------------------------
// block = 256 threads = 8 waves; each wave handles 32 edges (2 M-tiles).
// LDS: W1t[128][200], W2t[64][136], hid[8][32][136] f16  (~135 KB dynamic).
#define EW1_STRIDE 200
#define W2_STRIDE  136
#define HID_STRIDE 136

__global__ void __launch_bounds__(256)
edge_kernel(const float* __restrict__ x, const float* __restrict__ ea,
            const int* __restrict__ erow, const int* __restrict__ ecol,
            const float* __restrict__ W1, const float* __restrict__ b1,
            const float* __restrict__ W2, const float* __restrict__ b2,
            float* __restrict__ out_edge, float* __restrict__ agg) {
    extern __shared__ _Float16 smem[];
    _Float16* W1t = smem;                                   // 128 x 200
    _Float16* W2t = W1t + 128 * EW1_STRIDE;                 // 64 x 136
    _Float16* hid = W2t + 64 * W2_STRIDE;                   // 8 x 32 x 136

    const int tid = threadIdx.x;
    for (int i = tid; i < 192 * 128; i += 256) {            // W1t[n][k] = W1[k][n]
        int k = i >> 7, n = i & 127;
        W1t[n * EW1_STRIDE + k] = (_Float16)W1[i];
    }
    for (int i = tid; i < 128 * 64; i += 256) {             // W2t[n][k] = W2[k][n]
        int k = i >> 6, n = i & 63;
        W2t[n * W2_STRIDE + k] = (_Float16)W2[i];
    }
    __syncthreads();

    const int wid  = tid >> 5;
    const int lane = tid & 31;
    const int m    = lane & 15;
    const int h    = lane >> 4;
    const int e0   = (blockIdx.x * 8 + wid) * 32;   // 32-edge tile, exact fit

    // A-row sources for this lane's two M rows (m and m+16)
    const int em0 = e0 + m, em1 = e0 + 16 + m;
    const int rm0 = erow[em0], cm0 = ecol[em0];
    const int rm1 = erow[em1], cm1 = ecol[em1];
    const float* srcR0 = x  + (size_t)rm0 * LATENT;
    const float* srcC0 = x  + (size_t)cm0 * LATENT;
    const float* srcE0 = ea + (size_t)em0 * LATENT;
    const float* srcR1 = x  + (size_t)rm1 * LATENT;
    const float* srcC1 = x  + (size_t)cm1 * LATENT;
    const float* srcE1 = ea + (size_t)em1 * LATENT;

    // ---- GEMM1: [32x192] @ [192x128] ----
    v8f acc[2][8];
#pragma unroll
    for (int t = 0; t < 2; ++t)
#pragma unroll
        for (int nt = 0; nt < 8; ++nt) acc[t][nt] = (v8f)(0.0f);

#pragma unroll
    for (int kc = 0; kc < 6; ++kc) {
        const float* base0 = (kc < 2) ? srcR0 : (kc < 4) ? srcC0 : srcE0;
        const float* base1 = (kc < 2) ? srcR1 : (kc < 4) ? srcC1 : srcE1;
        int koff = (kc & 1) * 32;
        v16h a0 = frag_from_global_f32(base0, koff, h);
        v16h a1 = frag_from_global_f32(base1, koff, h);
#pragma unroll
        for (int nb = 0; nb < 2; ++nb) {
            v16h b[4];
#pragma unroll
            for (int j = 0; j < 4; ++j)
                b[j] = frag_from_lds(W1t, EW1_STRIDE, (nb * 4 + j) * 16 + m, kc * 32, h);
#pragma unroll
            for (int j = 0; j < 4; ++j) {
                acc[0][nb * 4 + j] = WMMA_F16(a0, b[j], acc[0][nb * 4 + j]);
                acc[1][nb * 4 + j] = WMMA_F16(a1, b[j], acc[1][nb * 4 + j]);
            }
        }
    }

    // bias + relu, restage hidden (f16) as A operand for GEMM2
    _Float16* hidW = hid + wid * 32 * HID_STRIDE;
#pragma unroll
    for (int t = 0; t < 2; ++t)
#pragma unroll
    for (int nt = 0; nt < 8; ++nt) {
        float bias = b1[nt * 16 + m];
#pragma unroll
        for (int r = 0; r < 8; ++r) {
            int mr = t * 16 + (h ? (r + 8) : r);
            float v = acc[t][nt][r] + bias;
            v = v > 0.0f ? v : 0.0f;
            hidW[mr * HID_STRIDE + nt * 16 + m] = (_Float16)v;
        }
    }
    // per-wave LDS RAW: DS ops are in-order within a wave; compiler adds waits.

    // ---- GEMM2: [32x128] @ [128x64] ----
    v8f acc2[2][4];
#pragma unroll
    for (int t = 0; t < 2; ++t)
#pragma unroll
        for (int nt = 0; nt < 4; ++nt) acc2[t][nt] = (v8f)(0.0f);

#pragma unroll
    for (int kc = 0; kc < 4; ++kc) {
        v16h a0 = frag_from_lds(hidW, HID_STRIDE, m, kc * 32, h);
        v16h a1 = frag_from_lds(hidW + 16 * HID_STRIDE, HID_STRIDE, m, kc * 32, h);
        v16h b[4];
#pragma unroll
        for (int nt = 0; nt < 4; ++nt)
            b[nt] = frag_from_lds(W2t, W2_STRIDE, nt * 16 + m, kc * 32, h);
#pragma unroll
        for (int nt = 0; nt < 4; ++nt) {
            acc2[0][nt] = WMMA_F16(a0, b[nt], acc2[0][nt]);
            acc2[1][nt] = WMMA_F16(a1, b[nt], acc2[1][nt]);
        }
    }

    // destination node of each row this lane touches in the C tiles
    int dst[2][8];
#pragma unroll
    for (int t = 0; t < 2; ++t)
#pragma unroll
        for (int r = 0; r < 8; ++r)
            dst[t][r] = ecol[e0 + t * 16 + (h ? (r + 8) : r)];

    // epilogue: residual add, write edge_attr_new, scatter-add into agg (L2)
#pragma unroll
    for (int t = 0; t < 2; ++t)
#pragma unroll
    for (int nt = 0; nt < 4; ++nt) {
        int   n    = nt * 16 + m;
        float bias = b2[n];
#pragma unroll
        for (int r = 0; r < 8; ++r) {
            int    e = e0 + t * 16 + (h ? (r + 8) : r);
            size_t o = (size_t)e * LATENT + n;
            float  v = ea[o] + bias + acc2[t][nt][r];
            out_edge[o] = v;
            atomicAdd(&agg[(size_t)dst[t][r] * LATENT + n], v);
        }
    }
}

// ---- node kernel ------------------------------------------------------------
// block = 256 threads = 8 waves; each wave handles 32 nodes (2 M-tiles).
// LDS: W1t[128][136], W2t[64][136], hid[8][32][136]  (~119 KB dynamic).
#define NW1_STRIDE 136

__global__ void __launch_bounds__(256)
node_kernel(const float* __restrict__ x, const float* __restrict__ agg,
            const float* __restrict__ W1, const float* __restrict__ b1,
            const float* __restrict__ W2, const float* __restrict__ b2,
            float* __restrict__ out_x) {
    extern __shared__ _Float16 smem[];
    _Float16* W1t = smem;                                   // 128 x 136
    _Float16* W2t = W1t + 128 * NW1_STRIDE;                 // 64 x 136
    _Float16* hid = W2t + 64 * W2_STRIDE;                   // 8 x 32 x 136

    const int tid = threadIdx.x;
    for (int i = tid; i < 128 * 128; i += 256) {
        int k = i >> 7, n = i & 127;
        W1t[n * NW1_STRIDE + k] = (_Float16)W1[i];
    }
    for (int i = tid; i < 128 * 64; i += 256) {
        int k = i >> 6, n = i & 63;
        W2t[n * W2_STRIDE + k] = (_Float16)W2[i];
    }
    __syncthreads();

    const int wid  = tid >> 5;
    const int lane = tid & 31;
    const int m    = lane & 15;
    const int h    = lane >> 4;
    const int n0   = (blockIdx.x * 8 + wid) * 32;
    if (n0 >= N_NODES) return;   // tail waves (no barriers past this point)

    const int nm0 = n0 + m, nm1 = n0 + 16 + m;
    const float* srcX0 = x   + (size_t)nm0 * LATENT;
    const float* srcA0 = agg + (size_t)nm0 * LATENT;
    const float* srcX1 = x   + (size_t)nm1 * LATENT;
    const float* srcA1 = agg + (size_t)nm1 * LATENT;

    // ---- GEMM1: [32x128] @ [128x128] ----
    v8f acc[2][8];
#pragma unroll
    for (int t = 0; t < 2; ++t)
#pragma unroll
        for (int nt = 0; nt < 8; ++nt) acc[t][nt] = (v8f)(0.0f);

#pragma unroll
    for (int kc = 0; kc < 4; ++kc) {
        const float* base0 = (kc < 2) ? srcX0 : srcA0;
        const float* base1 = (kc < 2) ? srcX1 : srcA1;
        int koff = (kc & 1) * 32;
        v16h a0 = frag_from_global_f32(base0, koff, h);
        v16h a1 = frag_from_global_f32(base1, koff, h);
#pragma unroll
        for (int nb = 0; nb < 2; ++nb) {
            v16h b[4];
#pragma unroll
            for (int j = 0; j < 4; ++j)
                b[j] = frag_from_lds(W1t, NW1_STRIDE, (nb * 4 + j) * 16 + m, kc * 32, h);
#pragma unroll
            for (int j = 0; j < 4; ++j) {
                acc[0][nb * 4 + j] = WMMA_F16(a0, b[j], acc[0][nb * 4 + j]);
                acc[1][nb * 4 + j] = WMMA_F16(a1, b[j], acc[1][nb * 4 + j]);
            }
        }
    }

    _Float16* hidW = hid + wid * 32 * HID_STRIDE;
#pragma unroll
    for (int t = 0; t < 2; ++t)
#pragma unroll
    for (int nt = 0; nt < 8; ++nt) {
        float bias = b1[nt * 16 + m];
#pragma unroll
        for (int r = 0; r < 8; ++r) {
            int mr = t * 16 + (h ? (r + 8) : r);
            float v = acc[t][nt][r] + bias;
            v = v > 0.0f ? v : 0.0f;
            hidW[mr * HID_STRIDE + nt * 16 + m] = (_Float16)v;
        }
    }

    // ---- GEMM2: [32x128] @ [128x64] ----
    v8f acc2[2][4];
#pragma unroll
    for (int t = 0; t < 2; ++t)
#pragma unroll
        for (int nt = 0; nt < 4; ++nt) acc2[t][nt] = (v8f)(0.0f);

#pragma unroll
    for (int kc = 0; kc < 4; ++kc) {
        v16h a0 = frag_from_lds(hidW, HID_STRIDE, m, kc * 32, h);
        v16h a1 = frag_from_lds(hidW + 16 * HID_STRIDE, HID_STRIDE, m, kc * 32, h);
        v16h b[4];
#pragma unroll
        for (int nt = 0; nt < 4; ++nt)
            b[nt] = frag_from_lds(W2t, W2_STRIDE, nt * 16 + m, kc * 32, h);
#pragma unroll
        for (int nt = 0; nt < 4; ++nt) {
            acc2[0][nt] = WMMA_F16(a0, b[nt], acc2[0][nt]);
            acc2[1][nt] = WMMA_F16(a1, b[nt], acc2[1][nt]);
        }
    }

#pragma unroll
    for (int t = 0; t < 2; ++t)
#pragma unroll
    for (int nt = 0; nt < 4; ++nt) {
        int   n    = nt * 16 + m;
        float bias = b2[n];
#pragma unroll
        for (int r = 0; r < 8; ++r) {
            int    node = n0 + t * 16 + (h ? (r + 8) : r);
            size_t o    = (size_t)node * LATENT + n;
            out_x[o] = x[o] + bias + acc2[t][nt][r];
        }
    }
}

// ---- launcher ---------------------------------------------------------------
extern "C" void kernel_launch(void* const* d_in, const int* in_sizes, int n_in,
                              void* d_out, int out_size, void* d_ws, size_t ws_size,
                              hipStream_t stream) {
    const float* x    = (const float*)d_in[0];
    const float* ea   = (const float*)d_in[1];
    const int*   eidx = (const int*)d_in[2];      // [2, N_EDGES]
    const float* eW1  = (const float*)d_in[3];
    const float* eb1  = (const float*)d_in[4];
    const float* eW2  = (const float*)d_in[5];
    const float* eb2  = (const float*)d_in[6];
    const float* nW1  = (const float*)d_in[7];
    const float* nb1  = (const float*)d_in[8];
    const float* nW2  = (const float*)d_in[9];
    const float* nb2  = (const float*)d_in[10];

    float* out_x    = (float*)d_out;                         // [N_NODES, 64]
    float* out_edge = out_x + (size_t)N_NODES * LATENT;      // [N_EDGES, 64]
    float* agg      = (float*)d_ws;                          // [N_NODES, 64] f32

    // 1) clear aggregation buffer (atomics accumulate; must zero each call)
    int aggN = N_NODES * LATENT;
    zero_kernel<<<(aggN + 255) / 256, 256, 0, stream>>>(agg, aggN);

    // 2) edge MLP + scatter-add (6250 blocks x 256 edges, exact)
    size_t smemE = (size_t)(128 * EW1_STRIDE + 64 * W2_STRIDE + 8 * 32 * HID_STRIDE)
                   * sizeof(_Float16);
    edge_kernel<<<N_EDGES / 256, 256, smemE, stream>>>(
        x, ea, eidx, eidx + N_EDGES, eW1, eb1, eW2, eb2, out_edge, agg);

    // 3) node MLP (3125 32-node tiles over 8 waves/block)
    size_t smemN = (size_t)(128 * NW1_STRIDE + 64 * W2_STRIDE + 8 * 32 * HID_STRIDE)
                   * sizeof(_Float16);
    int nodeTiles32 = (N_NODES + 31) / 32;                   // 3125 exact
    node_kernel<<<(nodeTiles32 + 7) / 8, 256, smemN, stream>>>(
        x, agg, nW1, nb1, nW2, nb2, out_x);
}
